// Network_58987080843722
// MI455X (gfx1250) — matrix operand order repaired and checked by reading, not compiled, not run on hardware
//
#include <hip/hip_runtime.h>
#include <hip/hip_bf16.h>

typedef __bf16 bf16_t;
typedef __attribute__((ext_vector_type(16))) __bf16 v16bf;
typedef __attribute__((ext_vector_type(8)))  __bf16 v8bf;
typedef __attribute__((ext_vector_type(8)))  float   v8f;

static constexpr int Kc = 128, Dc = 128;
static constexpr int Bb = 8, Pp = 8192;
static constexpr int MT = 32;          // points per workgroup tile
static constexpr int THREADS = 512;    // 16 waves (wave32)
static constexpr int NWAVES = 16;

#define WMMA_BF16(A, Bf, C) \
  __builtin_amdgcn_wmma_f32_16x16x32_bf16(false, (A), false, (Bf), (short)0, (C), false, false)

// ---------------------------------------------------------------------------
// Pack fp32 row-major (fin, fout) weights into bf16 WMMA-A tile order (W^T):
// A tile = 16(M=neuron) x 32(K); per ISA 16-bit A layout:
//   lane<16:  row m = lane,    K = {kb+0..7, kb+16..23},  kb = kt*32
//   lane>=16: row m = lane-16, K = {kb+8..15, kb+24..31}
// Each lane's 16 bf16 values stored contiguously (32 bytes -> 1 global load).
// K zero-padded to 32*Kt; N zero-padded (layer 6: fout 1 -> 16).
// ---------------------------------------------------------------------------
__global__ void pack_w(const float* __restrict__ W, bf16_t* __restrict__ out,
                       int fin, int freal, int Kt, int Mt) {
  int tid = blockIdx.x * blockDim.x + threadIdx.x;
  int total = Kt * Mt * 512;
  if (tid >= total) return;
  int j    = tid & 15;
  int lane = (tid >> 4) & 31;
  int tile = tid >> 9;
  int kt = tile % Kt;
  int mt = tile / Kt;
  int k = kt * 32 + ((lane >> 4) << 3) + ((j >> 3) << 4) + (j & 7);
  int n = mt * 16 + (lane & 15);               // output neuron
  float v = (k < fin && n < freal) ? W[(size_t)k * freal + n] : 0.0f;
  out[tid] = (bf16_t)v;
}

// Gather codes[indices[b]] (K=128 x D=128) and pack codes^T as WMMA-A (d rows).
__global__ void pack_codes(const float* __restrict__ codes,
                           const int* __restrict__ indices,
                           bf16_t* __restrict__ out) {
  int tid = blockIdx.x * blockDim.x + threadIdx.x;
  if (tid >= Bb * Kc * Dc) return;
  int j    = tid & 15;
  int lane = (tid >> 4) & 31;
  int tile = (tid >> 9) & 31;      // 32 tiles per batch (mt 0..7, kt 0..3)
  int b    = tid >> 14;            // 16384 elems per batch
  int kt = tile & 3;
  int mt = tile >> 2;
  int k = kt * 32 + ((lane >> 4) << 3) + ((j >> 3) << 4) + (j & 7);
  int d = mt * 16 + (lane & 15);
  size_t rec = (size_t)indices[b];
  out[tid] = (bf16_t)codes[(rec * Kc + k) * Dc + d];
}

// ---------------------------------------------------------------------------
// B fragment (activations): K x 16 points; per ISA 16-bit B layout:
//   lane<16:  n = lane,    K = kt*32 + {0..7, 16..23}
//   lane>=16: n = lane-16, K = kt*32 + {8..15, 24..31}
// ---------------------------------------------------------------------------
__device__ __forceinline__ v16bf load_b_frag(const bf16_t* buf, int stride,
                                             int p0, int kt, int lane) {
  const int p  = p0 + (lane & 15);
  const int kb = (kt << 5) + ((lane >> 4) << 3);
  const bf16_t* q = buf + p * stride + kb;
  v16bf bfr;
  ((v8bf*)&bfr)[0] = *(const v8bf*)(q);       // ds_load_b128
  ((v8bf*)&bfr)[1] = *(const v8bf*)(q + 16);  // ds_load_b128
  return bfr;
}

// Epilogue for one 16-neuron tile, both point halves: bias + leaky relu,
// pack to bf16, one contiguous ds_store_b128 per half.
template<bool RELU>
__device__ __forceinline__ void store_tile(bf16_t* __restrict__ outb, int ostride,
                                           const float* __restrict__ bias,
                                           int mt, int lane, v8f acc0, v8f acc1) {
  const int mb = (mt << 4) + ((lane >> 4) << 3);
  const v8f bv = *(const v8f*)(bias + mb);
  v8bf h0, h1;
#pragma unroll
  for (int r = 0; r < 8; ++r) {
    float v0 = acc0[r] + bv[r];
    float v1 = acc1[r] + bv[r];
    if (RELU) {
      v0 = v0 > 0.0f ? v0 : 0.02f * v0;
      v1 = v1 > 0.0f ? v1 : 0.02f * v1;
    }
    h0[r] = (bf16_t)v0;
    h1[r] = (bf16_t)v1;
  }
  *(v8bf*)(outb + (0  + (lane & 15)) * ostride + mb) = h0;
  *(v8bf*)(outb + (16 + (lane & 15)) * ostride + mb) = h1;
}

// One layer: out[p][n] = lrelu(sum_k in[p][k] * W[k][n] + b[n]).
// 2-way neuron-tile blocking: wave owns tiles (mt, mt+16); each B-fragment
// pair (both point halves) feeds 4 WMMAs -> halves LDS traffic per WMMA.
template<int KIN, int FOUT, int OSTRIDE, bool RELU>
__device__ __forceinline__ void mlp_layer(const bf16_t* __restrict__ inb,
                                          bf16_t* __restrict__ outb,
                                          const bf16_t* __restrict__ wp,
                                          const float* __restrict__ bias,
                                          int wave, int lane) {
  constexpr int Kt = KIN / 32;
  constexpr int Mt = FOUT / 16;
  int mt = wave;
  for (; mt + NWAVES < Mt; mt += 2 * NWAVES) {
    const bf16_t* w0 = wp + (size_t)(mt * Kt) * 512 + lane * 16;
    const bf16_t* w1 = wp + (size_t)((mt + NWAVES) * Kt) * 512 + lane * 16;
    v8f acc00 = {}, acc01 = {}, acc10 = {}, acc11 = {};
#pragma unroll 2
    for (int kt = 0; kt < Kt; ++kt) {
      v16bf a0 = *(const v16bf*)(w0 + (size_t)kt * 512);   // 32B global load
      v16bf a1 = *(const v16bf*)(w1 + (size_t)kt * 512);
      __builtin_prefetch(w0 + (size_t)(kt + 1) * 512, 0, 0);
      v16bf b0 = load_b_frag(inb, KIN, 0,  kt, lane);
      v16bf b1 = load_b_frag(inb, KIN, 16, kt, lane);
      acc00 = WMMA_BF16(a0, b0, acc00);
      acc01 = WMMA_BF16(a0, b1, acc01);
      acc10 = WMMA_BF16(a1, b0, acc10);
      acc11 = WMMA_BF16(a1, b1, acc11);
    }
    store_tile<RELU>(outb, OSTRIDE, bias, mt,          lane, acc00, acc01);
    store_tile<RELU>(outb, OSTRIDE, bias, mt + NWAVES, lane, acc10, acc11);
  }
  for (; mt < Mt; mt += NWAVES) {          // remainder: single tile
    const bf16_t* w0 = wp + (size_t)(mt * Kt) * 512 + lane * 16;
    v8f acc0 = {}, acc1 = {};
#pragma unroll 2
    for (int kt = 0; kt < Kt; ++kt) {
      v16bf a0 = *(const v16bf*)(w0 + (size_t)kt * 512);
      __builtin_prefetch(w0 + (size_t)(kt + 1) * 512, 0, 0);
      v16bf b0 = load_b_frag(inb, KIN, 0,  kt, lane);
      v16bf b1 = load_b_frag(inb, KIN, 16, kt, lane);
      acc0 = WMMA_BF16(a0, b0, acc0);
      acc1 = WMMA_BF16(a0, b1, acc1);
    }
    store_tile<RELU>(outb, OSTRIDE, bias, mt, lane, acc0, acc1);
  }
}

__device__ __forceinline__ void xcopy(bf16_t* dst, int stride, int fout,
                                      const bf16_t* xsave, int tid) {
  // append padded x (160 bf16 cols incl zero tail) after the fout h-columns
  for (int c = tid; c < 32 * 20; c += THREADS) {
    int row = c / 20, cc = (c % 20) * 8;
    *(v8bf*)(dst + row * stride + fout + cc) = *(const v8bf*)(xsave + row * 160 + cc);
  }
}

__global__ __launch_bounds__(THREADS) void fused_sdf(
    const int* __restrict__ indices,
    const float* __restrict__ qpts,        // (B,P,3)
    const float* __restrict__ cposg,       // (R,K,3)
    const bf16_t* __restrict__ w1p, const bf16_t* __restrict__ w2p,
    const bf16_t* __restrict__ w3p, const bf16_t* __restrict__ w4p,
    const bf16_t* __restrict__ w5p, const bf16_t* __restrict__ w6p,
    const bf16_t* __restrict__ codesp,     // (B, 32*512) packed codes^T
    const float* __restrict__ b1, const float* __restrict__ b2,
    const float* __restrict__ b3, const float* __restrict__ b4,
    const float* __restrict__ b5, const float* __restrict__ b6,
    float* __restrict__ out) {
  extern __shared__ char smem[];
  bf16_t* bufA  = (bf16_t*)smem;            // 32 x 2208  [point][col]
  bf16_t* bufB  = bufA + 32 * 2208;         // 32 x 1184
  bf16_t* xsave = bufB + 32 * 1184;         // 32 x 160
  float*  cpos  = (float*)(xsave + 5120);   // 128 x 3
  float*  qp    = cpos + 384;               // 32 x 3
  float*  wsum  = qp + 96;                  // 32
  float*  wtmp  = (float*)(bufA + 8192);    // 32x128 f32 overlay (interp only)
  bf16_t* wbf   = bufA;                     // 32x128 bf16 B-operand (interp only)

  const int tid  = threadIdx.x;
  const int wave = tid >> 5, lane = tid & 31;
  const int wg = blockIdx.x;
  const int b  = wg >> 8;                   // P/MT = 256 tiles per batch
  const int p0 = (wg & 255) << 5;

  const size_t rec = (size_t)indices[b];
  if (tid < 384) cpos[tid] = cposg[rec * 384 + tid];
  if (tid < 96)  qp[tid]   = qpts[((size_t)b * Pp + p0) * 3 + tid];
  __syncthreads();

  { // inverse-square-distance weights (DIST_SCALE=2 -> w = 1/sd)
    const int m  = tid >> 4;
    const int kb = (tid & 15) << 3;
    const float qx = qp[m * 3 + 0], qy = qp[m * 3 + 1], qz = qp[m * 3 + 2];
#pragma unroll
    for (int j = 0; j < 8; ++j) {
      const int k = kb + j;
      const float dx = qx - cpos[k * 3 + 0];
      const float dy = qy - cpos[k * 3 + 1];
      const float dz = qz - cpos[k * 3 + 2];
      wtmp[m * 128 + k] = 1.0f / (dx * dx + dy * dy + dz * dz + 1e-16f);
    }
  }
  __syncthreads();
  if (tid < 32) {
    float s = 0.0f;
    for (int k = 0; k < 128; ++k) s += wtmp[tid * 128 + k];
    wsum[tid] = 1.0f / s;
  }
  __syncthreads();
  {
    const int m  = tid >> 4;
    const int kb = (tid & 15) << 3;
    const float is = wsum[m];
#pragma unroll
    for (int j = 0; j < 8; ++j)
      wbf[m * 128 + kb + j] = (bf16_t)(wtmp[m * 128 + kb + j] * is);
  }
  __syncthreads();

  // interpolation GEMM: qc^T(128x32) = codes^T(128x128) @ w^T(128x32)
  // wave mt<8 owns a 16-dim tile, both point halves
  {
    const bf16_t* ca = codesp + (size_t)b * 16384;
    for (int mt = wave; mt < 8; mt += NWAVES) {
      const bf16_t* atile = ca + (size_t)(mt * 4) * 512 + lane * 16;
      v8f acc0 = {}, acc1 = {};
#pragma unroll
      for (int kt = 0; kt < 4; ++kt) {
        v16bf a  = *(const v16bf*)(atile + (size_t)kt * 512);
        v16bf c0 = load_b_frag(wbf, 128, 0,  kt, lane);
        v16bf c1 = load_b_frag(wbf, 128, 16, kt, lane);
        acc0 = WMMA_BF16(a, c0, acc0);
        acc1 = WMMA_BF16(a, c1, acc1);
      }
      const int mb = (mt << 4) + ((lane >> 4) << 3);   // dim base
      v8bf h0, h1;
#pragma unroll
      for (int r = 0; r < 8; ++r) { h0[r] = (bf16_t)acc0[r]; h1[r] = (bf16_t)acc1[r]; }
      *(v8bf*)(bufB + (0  + (lane & 15)) * 160 + mb) = h0;   // x cols 0..127
      *(v8bf*)(bufB + (16 + (lane & 15)) * 160 + mb) = h1;
    }
  }
  // x tail cols 128..130 = query point, 131..159 = zero pad
  for (int i = tid; i < 32 * 32; i += THREADS) {
    const int row = i >> 5, c = 128 + (i & 31);
    bf16_t v = (bf16_t)0.0f;
    if (c < 131) v = (bf16_t)qp[row * 3 + (c - 128)];
    bufB[row * 160 + c] = v;
  }
  __syncthreads();
  for (int i = tid; i < 5120; i += THREADS) xsave[i] = bufB[i];  // save x(160)
  __syncthreads();

  // decoder: ping-pong bufB(160) -> bufA(2208) -> bufB(1184) -> bufA(672)
  //          -> bufB(416) -> bufA(128) -> out
  mlp_layer< 160, 2048, 2208, true>(bufB, bufA, w1p, b1, wave, lane);
  __syncthreads();
  xcopy(bufA, 2208, 2048, xsave, tid);
  __syncthreads();
  mlp_layer<2208, 1024, 1184, true>(bufA, bufB, w2p, b2, wave, lane);
  __syncthreads();
  xcopy(bufB, 1184, 1024, xsave, tid);
  __syncthreads();
  mlp_layer<1184,  512,  672, true>(bufB, bufA, w3p, b3, wave, lane);
  __syncthreads();
  xcopy(bufA, 672, 512, xsave, tid);
  __syncthreads();
  mlp_layer< 672,  256,  416, true>(bufA, bufB, w4p, b4, wave, lane);
  __syncthreads();
  xcopy(bufB, 416, 256, xsave, tid);
  __syncthreads();
  mlp_layer< 416,  128,  128, true>(bufB, bufA, w5p, b5, wave, lane);
  __syncthreads();

  // final layer: neuron tile 0 only (fout 1 padded to 16); wave 0 does both
  // point halves; lanes 0..15 hold neuron 0 in acc[0]
  if (wave == 0) {
    const float b6v = b6[0];
    const bf16_t* wtile = w6p + lane * 16;
    v8f acc0 = {}, acc1 = {};
#pragma unroll
    for (int kt = 0; kt < 4; ++kt) {
      v16bf a  = *(const v16bf*)(wtile + (size_t)kt * 512);
      v16bf c0 = load_b_frag(bufA, 128, 0,  kt, lane);
      v16bf c1 = load_b_frag(bufA, 128, 16, kt, lane);
      acc0 = WMMA_BF16(a, c0, acc0);
      acc1 = WMMA_BF16(a, c1, acc1);
    }
    if (lane < 16) {
      float* o = out + (size_t)b * Pp + p0;
      o[lane]      = acc0[0] + b6v;
      o[lane + 16] = acc1[0] + b6v;
    }
  }
}

// ---------------------------------------------------------------------------
extern "C" void kernel_launch(void* const* d_in, const int* in_sizes, int n_in,
                              void* d_out, int out_size, void* d_ws, size_t ws_size,
                              hipStream_t stream) {
  (void)in_sizes; (void)n_in; (void)out_size; (void)ws_size;
  const int*   indices = (const int*)d_in[0];
  const float* qpts    = (const float*)d_in[1];
  const float* cposg   = (const float*)d_in[2];
  const float* codes   = (const float*)d_in[3];
  const float* W[6]; const float* Bv[6];
  for (int i = 0; i < 6; ++i) {
    W[i]  = (const float*)d_in[4 + 2 * i];
    Bv[i] = (const float*)d_in[5 + 2 * i];
  }

  // workspace layout (bf16 elems): packed W1..W6 then packed gathered codes
  static const size_t off[7] = {0, 327680, 2588672, 3194880, 3366912,
                                3420160, 3422208};
  bf16_t* ws = (bf16_t*)d_ws;
  bf16_t* wp[6];
  for (int i = 0; i < 6; ++i) wp[i] = ws + off[i];
  bf16_t* codesp = ws + off[6];   // + 131072 elems -> ~7.1 MB total

  static const int fin[6]   = {131, 2179, 1155, 643, 387, 128};
  static const int freal[6] = {2048, 1024, 512, 256, 128, 1};
  static const int Kt[6]    = {5, 69, 37, 21, 13, 4};
  static const int Mt[6]    = {128, 64, 32, 16, 8, 1};
  for (int i = 0; i < 6; ++i) {
    int total = Kt[i] * Mt[i] * 512;
    pack_w<<<(total + 511) / 512, 512, 0, stream>>>(W[i], wp[i], fin[i],
                                                    freal[i], Kt[i], Mt[i]);
  }
  pack_codes<<<(Bb * Kc * Dc) / 512, 512, 0, stream>>>(codes, indices, codesp);

  const size_t smem = 229632;   // bufA+bufB+xsave (bf16) + cpos/qp/wsum (f32)
  fused_sdf<<<Bb * (Pp / MT), THREADS, smem, stream>>>(
      indices, qpts, cposg,
      wp[0], wp[1], wp[2], wp[3], wp[4], wp[5], codesp,
      Bv[0], Bv[1], Bv[2], Bv[3], Bv[4], Bv[5],
      (float*)d_out);
}
